// LinkPrediction_61143154425965
// MI455X (gfx1250) — matrix operand order repaired and checked by reading, not compiled, not run on hardware
//
#include <hip/hip_runtime.h>

#define NNODES 100000
#define NEDGES 1600000
#define BATCH  4096

typedef __attribute__((ext_vector_type(16))) __bf16 v16bf;
typedef __attribute__((ext_vector_type(8)))  float  v8f;

__device__ __forceinline__ void atomic_add_f32(float* p, float v) {
  __hip_atomic_fetch_add(p, v, __ATOMIC_RELAXED, __HIP_MEMORY_SCOPE_AGENT);
}

// ---------------- utility kernels ----------------

__global__ void k_zero(float* __restrict__ p, int n) {
  int i = blockIdx.x * blockDim.x + threadIdx.x;
  if (i < n) p[i] = 0.0f;
}

__global__ void k_deg(const int* __restrict__ dst, float* __restrict__ deg, int ne) {
  int e = blockIdx.x * blockDim.x + threadIdx.x;
  if (e < ne) atomic_add_f32(&deg[dst[e]], 1.0f);
}

__global__ void k_invdeg(float* __restrict__ deg, int n) {
  int i = blockIdx.x * blockDim.x + threadIdx.x;
  if (i < n) deg[i] = 1.0f / fmaxf(deg[i], 1.0f);
}

// one wave per edge: lane l handles feature l and l+32
__global__ void k_scatter(const float* __restrict__ H, const int* __restrict__ src,
                          const int* __restrict__ dst, float* __restrict__ agg, int ne) {
  int e = blockIdx.x * (blockDim.x >> 5) + (threadIdx.x >> 5);
  if (e >= ne) return;
  int l = threadIdx.x & 31;
  long s = (long)src[e] * 64;
  long d = (long)dst[e] * 64;
  atomic_add_f32(&agg[d + l],      H[s + l]);
  atomic_add_f32(&agg[d + 32 + l], H[s + 32 + l]);
}

// ---------------- weight pre-packing into WMMA B-fragment layout (bf16) ----------------
// B 32x16 fragment layout: lane n = lane&15, khalf = lane>>4, elem e<8 -> K = khalf*8 + e, e>=8 -> K = 16 + khalf*8 + (e-8).
// pack[f*32 + lane] holds the 16 bf16 elems for fragment f = kb*4 + nt.

// fused [Ws;Wn] (K=128 -> 4 k-blocks x 4 n-tiles = 16 fragments, 512 threads)
__global__ void k_pack_sage(const float* __restrict__ Ws, const float* __restrict__ Wn,
                            v16bf* __restrict__ pack) {
  int t = blockIdx.x * blockDim.x + threadIdx.x;
  if (t >= 512) return;
  int lane = t & 31, f = t >> 5;
  int kb = f >> 2, nt = f & 3;
  int n = nt * 16 + (lane & 15);
  int khalf = lane >> 4;
  v16bf b;
#pragma unroll
  for (int e = 0; e < 16; ++e) {
    int K = kb * 32 + (e < 8 ? khalf * 8 + e : 16 + khalf * 8 + (e - 8));
    float w = (K < 64) ? Ws[K * 64 + n] : Wn[(K - 64) * 64 + n];
    b[e] = (__bf16)w;
  }
  pack[f * 32 + lane] = b;
}

// single 64x64 weight (K=64 -> 2 k-blocks x 4 n-tiles = 8 fragments, 256 threads)
__global__ void k_pack64(const float* __restrict__ W, v16bf* __restrict__ pack) {
  int t = blockIdx.x * blockDim.x + threadIdx.x;
  if (t >= 256) return;
  int lane = t & 31, f = t >> 5;
  int kb = f >> 2, nt = f & 3;
  int n = nt * 16 + (lane & 15);
  int khalf = lane >> 4;
  v16bf b;
#pragma unroll
  for (int e = 0; e < 16; ++e) {
    int K = kb * 32 + (e < 8 ? khalf * 8 + e : 16 + khalf * 8 + (e - 8));
    b[e] = (__bf16)W[K * 64 + n];
  }
  pack[f * 32 + lane] = b;
}

// ---------------- A fragment: vectorized f32 loads + cvt to bf16 ----------------
// A 16x32: lane m = lane&15; khalf = lane>>4; elems 0..7 -> K=kbase+khalf*8+e,
//          elems 8..15 -> K=kbase+16+khalf*8+(e-8). Each 8-elem group is contiguous f32.
__device__ __forceinline__ v16bf load_afrag(const float* __restrict__ row,
                                            int khalf, int kbase, float scale) {
  const float4* p = (const float4*)(row + kbase + khalf * 8);
  const float4* q = (const float4*)(row + kbase + 16 + khalf * 8);
  float4 x0 = p[0], x1 = p[1];
  float4 y0 = q[0], y1 = q[1];
  v16bf a;
  a[0]  = (__bf16)(x0.x * scale); a[1]  = (__bf16)(x0.y * scale);
  a[2]  = (__bf16)(x0.z * scale); a[3]  = (__bf16)(x0.w * scale);
  a[4]  = (__bf16)(x1.x * scale); a[5]  = (__bf16)(x1.y * scale);
  a[6]  = (__bf16)(x1.z * scale); a[7]  = (__bf16)(x1.w * scale);
  a[8]  = (__bf16)(y0.x * scale); a[9]  = (__bf16)(y0.y * scale);
  a[10] = (__bf16)(y0.z * scale); a[11] = (__bf16)(y0.w * scale);
  a[12] = (__bf16)(y1.x * scale); a[13] = (__bf16)(y1.y * scale);
  a[14] = (__bf16)(y1.z * scale); a[15] = (__bf16)(y1.w * scale);
  return a;
}

// ---------------- fused SAGE layer GEMM: out = act(Hin@Ws + (Agg*invdeg)@Wn + b) ----------------
// one wave (32 threads) per 16-row block; ACT: 0=elu, 1=relu
template <int ACT>
__global__ void __launch_bounds__(32)
k_sage_gemm(const float* __restrict__ Hin, const float* __restrict__ Agg,
            const float* __restrict__ invdeg, const v16bf* __restrict__ packB,
            const float* __restrict__ bias, float* __restrict__ Hout) {
  const int lane  = threadIdx.x & 31;
  const int m     = lane & 15;
  const int khalf = lane >> 4;
  const int rb    = blockIdx.x;
  const long arow = (long)rb * 16 + m;

  const float* hrow = Hin + arow * 64;
  const float* grow = Agg + arow * 64;
  const float  idg  = invdeg[arow];

  v16bf A[4];
  A[0] = load_afrag(hrow, khalf, 0, 1.0f);
  A[1] = load_afrag(hrow, khalf, 32, 1.0f);
  A[2] = load_afrag(grow, khalf, 0, idg);
  A[3] = load_afrag(grow, khalf, 32, idg);

#pragma unroll
  for (int nt = 0; nt < 4; ++nt) {
    const int n = nt * 16 + m;
    v8f acc = {};
#pragma unroll
    for (int kb = 0; kb < 4; ++kb) {
      v16bf B = packB[(kb * 4 + nt) * 32 + lane];
      acc = __builtin_amdgcn_wmma_f32_16x16x32_bf16(false, A[kb], false, B, (short)0, acc,
                                                    false, false);
    }
    const float bcol = bias[n];
#pragma unroll
    for (int r = 0; r < 8; ++r) {
      long orow = (long)rb * 16 + khalf * 8 + r;
      float v = acc[r] + bcol;
      if (ACT == 0) v = (v > 0.0f) ? v : expm1f(v);  // elu
      if (ACT == 1) v = fmaxf(v, 0.0f);              // relu
      Hout[orow * 64 + n] = v;
    }
  }
}

// ---------------- generic M x 64 @ 64 x 64 GEMM: C = act(A@W + bias); ACT: 1=relu, 2=none -------
template <int ACT>
__global__ void __launch_bounds__(32)
k_gemm64(const float* __restrict__ Ain, const v16bf* __restrict__ packB,
         const float* __restrict__ bias, float* __restrict__ Cout) {
  const int lane  = threadIdx.x & 31;
  const int m     = lane & 15;
  const int khalf = lane >> 4;
  const int rb    = blockIdx.x;
  const float* arow = Ain + ((long)rb * 16 + m) * 64;

  v16bf A0 = load_afrag(arow, khalf, 0, 1.0f);
  v16bf A1 = load_afrag(arow, khalf, 32, 1.0f);

#pragma unroll
  for (int nt = 0; nt < 4; ++nt) {
    const int n = nt * 16 + m;
    v8f acc = {};
    v16bf B0 = packB[(0 * 4 + nt) * 32 + lane];
    v16bf B1 = packB[(1 * 4 + nt) * 32 + lane];
    acc = __builtin_amdgcn_wmma_f32_16x16x32_bf16(false, A0, false, B0, (short)0, acc, false, false);
    acc = __builtin_amdgcn_wmma_f32_16x16x32_bf16(false, A1, false, B1, (short)0, acc, false, false);
    const float bcol = bias[n];
#pragma unroll
    for (int r = 0; r < 8; ++r) {
      long orow = (long)rb * 16 + khalf * 8 + r;
      float v = acc[r] + bcol;
      if (ACT == 1) v = fmaxf(v, 0.0f);
      Cout[orow * 64 + n] = v;
    }
  }
}

// ---------------- column mean of Y (nrows x 64) into ym[64] (ym must be zeroed) ----------------
__global__ void k_colmean(const float* __restrict__ Y, float* __restrict__ ym, int nrows) {
  __shared__ float sdata[256];
  const int col = threadIdx.x & 63;
  const int sub = threadIdx.x >> 6;  // 0..3
  float s = 0.0f;
  for (long row = (long)blockIdx.x * 4 + sub; row < nrows; row += (long)gridDim.x * 4)
    s += Y[row * 64 + col];
  sdata[threadIdx.x] = s;
  __syncthreads();
  if (threadIdx.x < 64) {
    float t = sdata[threadIdx.x] + sdata[threadIdx.x + 64] +
              sdata[threadIdx.x + 128] + sdata[threadIdx.x + 192];
    atomic_add_f32(&ym[col], t * (1.0f / (float)NNODES));
  }
}

// wvec[n] = bl[n] + sum_k ym[k] * Wl[(64+k)*64 + n]   (the broadcast-ym half of the concat GEMM)
__global__ void k_wvec(const float* __restrict__ ym, const float* __restrict__ Wl,
                       const float* __restrict__ bl, float* __restrict__ wvec) {
  int n = threadIdx.x;
  if (n < 64) {
    float s = bl[n];
#pragma unroll 4
    for (int k = 0; k < 64; ++k) s += ym[k] * Wl[(64 + k) * 64 + n];
    wvec[n] = s;
  }
}

// ---------------- launcher ----------------

extern "C" void kernel_launch(void* const* d_in, const int* in_sizes, int n_in,
                              void* d_out, int out_size, void* d_ws, size_t ws_size,
                              hipStream_t stream) {
  (void)in_sizes; (void)n_in; (void)out_size; (void)ws_size;

  const float* node = (const float*)d_in[0];
  const float* X    = (const float*)d_in[1];
  const int*   ei   = (const int*)  d_in[2];
  const float* Wi   = (const float*)d_in[3];
  const float* bi   = (const float*)d_in[4];
  const float* Ws0  = (const float*)d_in[5];
  const float* Wn0  = (const float*)d_in[6];
  const float* b0   = (const float*)d_in[7];
  const float* Ws1  = (const float*)d_in[8];
  const float* Wn1  = (const float*)d_in[9];
  const float* b1   = (const float*)d_in[10];
  const float* Ws2  = (const float*)d_in[11];
  const float* Wn2  = (const float*)d_in[12];
  const float* b2   = (const float*)d_in[13];
  const float* Wl   = (const float*)d_in[14];
  const float* bl   = (const float*)d_in[15];
  const float* Wo   = (const float*)d_in[16];
  const float* bo   = (const float*)d_in[17];

  const int* src = ei;
  const int* dst = ei + NEDGES;

  const size_t NF = (size_t)NNODES * 64;  // 6,400,000 floats
  float* bufA = (float*)d_ws;             // agg scratch
  float* bufB = bufA + NF;                // h ping
  float* bufC = bufB + NF;                // h pong
  float* deg  = bufC + NF;                // NNODES (deg -> invdeg in place)
  float* ym   = deg + NNODES;             // 64
  float* wvec = ym + 64;                  // 64
  float* xbuf = wvec + 64;                // BATCH*64
  float* zbuf = xbuf + (size_t)BATCH * 64;
  // packed bf16 weight fragments (32B-aligned: preceding float count is a multiple of 8)
  v16bf* pS0 = (v16bf*)(zbuf + (size_t)BATCH * 64);  // 512 frags-lanes * 32B = 16 KB
  v16bf* pS1 = pS0 + 512;
  v16bf* pS2 = pS1 + 512;
  v16bf* pWi = pS2 + 512;                            // 256 * 32B = 8 KB each
  v16bf* pWl = pWi + 256;
  v16bf* pWo = pWl + 256;

  const int T = 256;
  const int nfInt = (int)NF;

  // pack all weights to bf16 WMMA fragments (tiny, once per call)
  k_pack_sage<<<2, T, 0, stream>>>(Ws0, Wn0, pS0);
  k_pack_sage<<<2, T, 0, stream>>>(Ws1, Wn1, pS1);
  k_pack_sage<<<2, T, 0, stream>>>(Ws2, Wn2, pS2);
  k_pack64  <<<1, T, 0, stream>>>(Wi, pWi);
  k_pack64  <<<1, T, 0, stream>>>(Wl, pWl);  // top 64 rows of Wl
  k_pack64  <<<1, T, 0, stream>>>(Wo, pWo);

  // degrees (once for all three layers)
  k_zero  <<<(NNODES + T - 1) / T, T, 0, stream>>>(deg, NNODES);
  k_deg   <<<(NEDGES + T - 1) / T, T, 0, stream>>>(dst, deg, NEDGES);
  k_invdeg<<<(NNODES + T - 1) / T, T, 0, stream>>>(deg, NNODES);

  const int rbN = NNODES / 16;  // 6250
  const int rbB = BATCH  / 16;  // 256
  const int scatBlocks = (NEDGES + 7) / 8;  // 8 waves/block, 1 edge/wave

  // layer 0: elu(X@Ws0 + mean@Wn0 + b0) -> bufB
  k_zero        <<<(nfInt + T - 1) / T, T, 0, stream>>>(bufA, nfInt);
  k_scatter     <<<scatBlocks, T, 0, stream>>>(X, src, dst, bufA, NEDGES);
  k_sage_gemm<0><<<rbN, 32, 0, stream>>>(X, bufA, deg, pS0, b0, bufB);

  // layer 1: relu -> bufC
  k_zero        <<<(nfInt + T - 1) / T, T, 0, stream>>>(bufA, nfInt);
  k_scatter     <<<scatBlocks, T, 0, stream>>>(bufB, src, dst, bufA, NEDGES);
  k_sage_gemm<1><<<rbN, 32, 0, stream>>>(bufB, bufA, deg, pS1, b1, bufC);

  // layer 2: relu -> bufB (= y)
  k_zero        <<<(nfInt + T - 1) / T, T, 0, stream>>>(bufA, nfInt);
  k_scatter     <<<scatBlocks, T, 0, stream>>>(bufC, src, dst, bufA, NEDGES);
  k_sage_gemm<1><<<rbN, 32, 0, stream>>>(bufC, bufA, deg, pS2, b2, bufB);

  // mean pool + fold ym half of Wl into a bias vector
  k_zero   <<<1, 64, 0, stream>>>(ym, 64);
  k_colmean<<<1024, T, 0, stream>>>(bufB, ym, NNODES);
  k_wvec   <<<1, 64, 0, stream>>>(ym, Wl, bl, wvec);

  // head: x = node@Wi + bi ; z = relu(x@Wl_top + wvec) ; out = z@Wo + bo
  k_gemm64<2><<<rbB, 32, 0, stream>>>(node, pWi, bi, xbuf);
  k_gemm64<1><<<rbB, 32, 0, stream>>>(xbuf, pWl, wvec, zbuf);
  k_gemm64<2><<<rbB, 32, 0, stream>>>(zbuf, pWo, bo, (float*)d_out);
}
